// LocalGraph_74079595921839
// MI455X (gfx1250) — compile-verified, matrix operand order checked by reading, not run on hardware
//
#include <hip/hip_runtime.h>
#include <hip/hip_fp16.h>
#include <math.h>

#define D 64
#define TILE_M 16
#define SEED_NUM 4096
#define EPS_DIV 1e-8f
#define EPS_NORM 1e-12f

typedef _Float16 half_t;
typedef _Float16 v16h __attribute__((ext_vector_type(16)));
typedef float    v8f  __attribute__((ext_vector_type(8)));

// HW float atomic (global_atomic_add_f32); L2-resident on MI455X (192MB L2).
__device__ __forceinline__ void atomic_add_f32(float* p, float v) {
    unsafeAtomicAdd(p, v);
}

// ---------------------------------------------------------------- utilities
__global__ void zero_kernel(float* p, size_t n) {
    size_t i = (size_t)blockIdx.x * blockDim.x + threadIdx.x;
    size_t stride = (size_t)gridDim.x * blockDim.x;
    for (; i < n; i += stride) p[i] = 0.0f;
}

// row degree: order[n] = #edges with rows==n
__global__ void degree_kernel(const int* __restrict__ rows, float* deg, int E) {
    int e = blockIdx.x * blockDim.x + threadIdx.x;
    if (e < E) atomic_add_f32(&deg[rows[e]], 1.0f);
}

// acc[rows[e]] += x[cols[e]]   (16 lanes per edge, float4 each)
__global__ void spmm_scatter_kernel(const int* __restrict__ rows,
                                    const int* __restrict__ cols,
                                    const float* __restrict__ x,
                                    float* acc, int E) {
    int gid  = blockIdx.x * blockDim.x + threadIdx.x;
    int e    = gid >> 4;
    int lane = gid & 15;
    if (e >= E) return;
    int r = rows[e], c = cols[e];
    const float4 v = *reinterpret_cast<const float4*>(&x[(size_t)c * D + lane * 4]);
    float* dst = &acc[(size_t)r * D + lane * 4];
    atomic_add_f32(dst + 0, v.x);
    atomic_add_f32(dst + 1, v.y);
    atomic_add_f32(dst + 2, v.z);
    atomic_add_f32(dst + 3, v.w);
}

// acc1 -= emb  (turns spmm(emb) into fst_embeds, in place)
__global__ void sub_inplace_kernel(float* a, const float* __restrict__ b, size_t n) {
    size_t i = (size_t)blockIdx.x * blockDim.x + threadIdx.x;
    size_t stride = (size_t)gridDim.x * blockDim.x;
    for (; i < n; i += stride) a[i] -= b[i];
}

// second hop: acc2[rows[e]] += fst[cols[e]]; scdnum[rows[e]] += deg[cols[e]]
__global__ void spmm2_kernel(const int* __restrict__ rows,
                             const int* __restrict__ cols,
                             const float* __restrict__ fst,
                             const float* __restrict__ deg,
                             float* acc2, float* scdnum, int E) {
    int gid  = blockIdx.x * blockDim.x + threadIdx.x;
    int e    = gid >> 4;
    int lane = gid & 15;
    if (e >= E) return;
    int r = rows[e], c = cols[e];
    const float4 v = *reinterpret_cast<const float4*>(&fst[(size_t)c * D + lane * 4]);
    float* dst = &acc2[(size_t)r * D + lane * 4];
    atomic_add_f32(dst + 0, v.x);
    atomic_add_f32(dst + 1, v.y);
    atomic_add_f32(dst + 2, v.z);
    atomic_add_f32(dst + 3, v.w);
    if (lane == 0) atomic_add_f32(&scdnum[r], deg[c]);
}

// -------------------------------------------------------- WMMA score kernel
// One wave32 per 16-row tile. Algebra:
//   fst+scd          = acc2 - deg*emb           (acc2 = spmm(fst) here)
//   fst_num+scd_num  = scdnum - deg
//   subgraph         = (acc2 - deg*emb) / (scdnum - deg + eps)
// Then l2-normalize rows of subgraph and emb, batched dot via
// v_wmma_f32_16x16x32_f16 (diag of 16x16 D-tile), sigmoid/log/gumbel.
__global__ void __launch_bounds__(32)
score_wmma_kernel(const float* __restrict__ emb,
                  const float* __restrict__ acc2,
                  const float* __restrict__ deg,
                  const float* __restrict__ scdnum,
                  float* scores_out, int N) {
    __shared__ float  s_emb[TILE_M][D];
    __shared__ float  s_sub[TILE_M][D];
    __shared__ half_t h_emb[TILE_M][D];
    __shared__ half_t h_sub[TILE_M][D];
    __shared__ float  s_a[TILE_M];       // reused: deg, then 1/||emb||
    __shared__ float  s_b[TILE_M];       // reused: denom, then 1/||sub||
    __shared__ float  dtile[TILE_M][TILE_M + 1];

    const int lane = threadIdx.x;
    const int rb   = blockIdx.x * TILE_M;
    const bool full = (rb + TILE_M) <= N;

    if (full) {
        // Async global->LDS staging: each lane copies 32 floats (8 x b128)
        // of both emb and acc2 tiles. Generic LDS pointers truncate to the
        // wave-relative LDS byte address (low 32 bits).
        const int r  = lane >> 1;
        const int c0 = (lane & 1) * 32;
        unsigned lds_e = (unsigned)(size_t)&s_emb[r][c0];
        unsigned lds_a = (unsigned)(size_t)&s_sub[r][c0];
        unsigned voff  = (unsigned)((r * D + c0) * sizeof(float));
        const float* base_e = emb  + (size_t)rb * D;
        const float* base_a = acc2 + (size_t)rb * D;
#pragma unroll
        for (int q = 0; q < 8; ++q) {
            asm volatile("global_load_async_to_lds_b128 %0, %1, %2 offset:0"
                         :: "v"(lds_e + q * 16u), "v"(voff + q * 16u), "s"(base_e)
                         : "memory");
            asm volatile("global_load_async_to_lds_b128 %0, %1, %2 offset:0"
                         :: "v"(lds_a + q * 16u), "v"(voff + q * 16u), "s"(base_a)
                         : "memory");
        }
        asm volatile("s_wait_asynccnt 0" ::: "memory");
    } else {
        for (int t = lane; t < TILE_M * D; t += 32) {
            int r = t >> 6, c = t & 63;
            int gr = min(rb + r, N - 1);
            s_emb[r][c] = emb [(size_t)gr * D + c];
            s_sub[r][c] = acc2[(size_t)gr * D + c];
        }
    }
    if (lane < TILE_M) {
        int gr = min(rb + lane, N - 1);
        float dg = deg[gr];
        s_a[lane] = dg;
        s_b[lane] = scdnum[gr] - dg + EPS_DIV;
    }
    __syncthreads();

    // subgraph = (acc2 - deg*emb) / denom
    for (int t = lane; t < TILE_M * D; t += 32) {
        int r = t >> 6, c = t & 63;
        s_sub[r][c] = (s_sub[r][c] - s_a[r] * s_emb[r][c]) / s_b[r];
    }
    __syncthreads();

    // per-row L2 norms
    if (lane < TILE_M) {
        float se = 0.0f, ss = 0.0f;
        for (int c = 0; c < D; ++c) {
            float a = s_emb[lane][c], b = s_sub[lane][c];
            se += a * a; ss += b * b;
        }
        s_a[lane] = 1.0f / fmaxf(sqrtf(se), EPS_NORM);
        s_b[lane] = 1.0f / fmaxf(sqrtf(ss), EPS_NORM);
    }
    __syncthreads();

    // normalized values are in [-1,1] -> fp16 fragments are safe
    for (int t = lane; t < TILE_M * D; t += 32) {
        int r = t >> 6, c = t & 63;
        h_emb[r][c] = (half_t)(s_emb[r][c] * s_a[r]);
        h_sub[r][c] = (half_t)(s_sub[r][c] * s_b[r]);
    }
    __syncthreads();

    // Build A (16x32 f16, rows = subgraph) and B (32x16 f16, cols = emb rows)
    // fragments per the documented CDNA5 16-bit layout:
    //   lane&15 -> M (or N); lane>>4 selects K {0..7,16..23} vs {8..15,24..31}
    const int m  = lane & 15;
    const int hi = lane >> 4;
    v8f c_acc = {};
#pragma unroll
    for (int k0 = 0; k0 < D; k0 += 32) {
        v16h a, b;
#pragma unroll
        for (int e2 = 0; e2 < 16; ++e2) {
            int v = e2 >> 1;
            int k = ((v < 4) ? (hi * 8 + 2 * v) : (16 + hi * 8 + 2 * (v - 4))) + (e2 & 1);
            a[e2] = h_sub[m][k0 + k];
            b[e2] = h_emb[m][k0 + k];
        }
        c_acc = __builtin_amdgcn_wmma_f32_16x16x32_f16(
            false, a, false, b, (short)0, c_acc, false, false);
    }

    // scatter 16x16 D-tile to LDS (lane<16: N=lane, M=vgpr; lane>=16: M=vgpr+8)
#pragma unroll
    for (int j = 0; j < 8; ++j) dtile[hi * 8 + j][m] = c_acc[j];
    __syncthreads();

    if (lane < TILE_M && (rb + lane) < N) {
        float dot = dtile[lane][lane];
        float sc  = 1.0f / (1.0f + __expf(-dot));
        // deterministic hash-based Gumbel noise (stand-in for jax key 42)
        unsigned idx = rb + lane;
        unsigned h = idx * 747796405u + 2891336453u;
        h ^= h >> 17; h *= 0xed5ad4bbu; h ^= h >> 11; h *= 0xac4c1b51u; h ^= h >> 15;
        float u = (h >> 8) * (1.0f / 16777216.0f);
        u = fminf(fmaxf(u, 1e-7f), 1.0f - 1e-7f);
        scores_out[idx] = __logf(sc) - __logf(-__logf(u));
    }
}

// ------------------------------------------------------------------- top-k
__device__ __forceinline__ unsigned sortable_key(float f) {
    unsigned u = __float_as_uint(f);
    return (u & 0x80000000u) ? ~u : (u | 0x80000000u);
}

__global__ void hist_kernel(const float* __restrict__ scores, unsigned* hist, int N) {
    int i = blockIdx.x * blockDim.x + threadIdx.x;
    if (i < N) atomicAdd(&hist[sortable_key(scores[i]) >> 24], 1u);
}

__global__ void thresh_kernel(const unsigned* __restrict__ hist, unsigned* meta) {
    if (threadIdx.x == 0 && blockIdx.x == 0) {
        unsigned cum = 0; int b = 255;
        for (; b >= 0; --b) {
            if (cum + hist[b] >= SEED_NUM) break;
            cum += hist[b];
        }
        if (b < 0) b = 0;
        meta[0] = (unsigned)b;  // threshold bucket
        meta[1] = cum;          // count strictly above it
        meta[2] = 0;            // counter: above-bucket slots
        meta[3] = 0;            // counter: in-bucket slots
    }
}

__global__ void collect_kernel(const float* __restrict__ scores,
                               unsigned* meta, float* seeds, int N) {
    int i = blockIdx.x * blockDim.x + threadIdx.x;
    if (i >= N) return;
    unsigned b   = sortable_key(scores[i]) >> 24;
    unsigned thr = meta[0];
    if (b > thr) {
        unsigned slot = atomicAdd(&meta[2], 1u);
        if (slot < SEED_NUM) seeds[slot] = (float)i;
    } else if (b == thr) {
        unsigned slot = meta[1] + atomicAdd(&meta[3], 1u);
        if (slot < SEED_NUM) seeds[slot] = (float)i;
    }
}

// ------------------------------------------------------------------ launch
extern "C" void kernel_launch(void* const* d_in, const int* in_sizes, int n_in,
                              void* d_out, int out_size, void* d_ws, size_t ws_size,
                              hipStream_t stream) {
    const int*   rows = (const int*)d_in[0];
    const int*   cols = (const int*)d_in[1];
    const float* emb  = (const float*)d_in[2];
    const int E = in_sizes[0];
    const int N = in_sizes[2] / D;

    float* ws   = (float*)d_ws;
    float* acc1 = ws;                               // N*D : spmm(emb) -> fst
    float* acc2 = acc1 + (size_t)N * D;             // N*D : spmm(fst)
    float* deg  = acc2 + (size_t)N * D;             // N
    float* scdn = deg + N;                          // N   : spmm(deg)
    unsigned* hist = (unsigned*)(scdn + N);         // 256
    unsigned* meta = hist + 256;                    // 4

    float* scores = (float*)d_out;                  // [N]
    float* seeds  = scores + N;                     // [SEED_NUM]

    size_t zn = 2 * (size_t)N * D + 2 * (size_t)N + 256 + 4;
    zero_kernel<<<1024, 256, 0, stream>>>(ws, zn);

    degree_kernel<<<(E + 255) / 256, 256, 0, stream>>>(rows, deg, E);

    int spmm_blocks = (E * 16 + 255) / 256;
    spmm_scatter_kernel<<<spmm_blocks, 256, 0, stream>>>(rows, cols, emb, acc1, E);

    sub_inplace_kernel<<<1024, 256, 0, stream>>>(acc1, emb, (size_t)N * D);

    spmm2_kernel<<<spmm_blocks, 256, 0, stream>>>(rows, cols, acc1, deg, acc2, scdn, E);

    score_wmma_kernel<<<(N + TILE_M - 1) / TILE_M, 32, 0, stream>>>(
        emb, acc2, deg, scdn, scores, N);

    hist_kernel<<<(N + 255) / 256, 256, 0, stream>>>(scores, hist, N);
    thresh_kernel<<<1, 1, 0, stream>>>(hist, meta);
    collect_kernel<<<(N + 255) / 256, 256, 0, stream>>>(scores, meta, seeds, N);
}
